// MultiHeadAttentionBlock_12713103196400
// MI455X (gfx1250) — matrix operand order repaired and checked by reading, not compiled, not run on hardware
//
#include <hip/hip_runtime.h>

// Problem constants (from reference): B=4, H=16, S=2048, DK=64
#define B_   4
#define H_   16
#define S_   2048
#define DK_  64
#define SP_  2052          // padded LDS row stride in floats (2048 + 4): kills bank conflicts
#define SCALE_ 0.125f      // 1/sqrt(64)

typedef __attribute__((ext_vector_type(16))) __bf16 v16bf;
typedef __attribute__((ext_vector_type(4)))  __bf16 v4bf;
typedef __attribute__((ext_vector_type(8)))  float  v8f;

__device__ __forceinline__ __bf16 f32_to_bf16_rne(float f) {
    unsigned u = __builtin_bit_cast(unsigned, f);
    unsigned r = u + 0x7FFFu + ((u >> 16) & 1u);   // round-to-nearest-even
    unsigned short hv = (unsigned short)(r >> 16);
    return __builtin_bit_cast(__bf16, hv);
}
__device__ __forceinline__ float bf16_to_f32(__bf16 h) {
    unsigned u = ((unsigned)__builtin_bit_cast(unsigned short, h)) << 16;
    return __builtin_bit_cast(float, u);
}
__device__ __forceinline__ void split_bf16(float f, __bf16& hi, __bf16& lo) {
    __bf16 h = f32_to_bf16_rne(f);
    hi = h;
    lo = f32_to_bf16_rne(f - bf16_to_f32(h));
}

// ---------------------------------------------------------------------------
// Pre-pass: one-time split of K (fp32) into bf16 hi/lo planes in workspace.
// 8.39M elements; runs once, keeps all conversion VALU out of the matmul loop.
// ---------------------------------------------------------------------------
__global__ __launch_bounds__(256)
void convert_k_kernel(const float* __restrict__ K,
                      unsigned short* __restrict__ Khi,
                      unsigned short* __restrict__ Klo) {
    const size_t i = ((size_t)blockIdx.x * 256 + threadIdx.x) * 4;
    float4 v = *(const float4*)(K + i);
    v4bf h, l;
    __bf16 hh, ll;
    split_bf16(v.x, hh, ll); h[0] = hh; l[0] = ll;
    split_bf16(v.y, hh, ll); h[1] = hh; l[1] = ll;
    split_bf16(v.z, hh, ll); h[2] = hh; l[2] = ll;
    split_bf16(v.w, hh, ll); h[3] = hh; l[3] = ll;
    *(v4bf*)(Khi + i) = h;
    *(v4bf*)(Klo + i) = l;
}

// ---------------------------------------------------------------------------
// Main kernel: one workgroup = 16 query rows of one (b,h); 8 waves, each wave
// owns 256 key columns. Scores accumulate via 3xBF16 WMMA into LDS, then row
// softmax and a single coalesced fp32 store (the only HBM-heavy traffic).
// ---------------------------------------------------------------------------
template <bool PRECONV>
__global__ __launch_bounds__(256)
void attn_softmax_kernel(const float* __restrict__ Q,
                         const float* __restrict__ K,
                         const unsigned short* __restrict__ Khi,
                         const unsigned short* __restrict__ Klo,
                         float* __restrict__ Out) {
    extern __shared__ float sc[];              // [16][SP_] score buffer (131,328 B)

    const int lane = threadIdx.x & 31;
    const int wave = threadIdx.x >> 5;         // 0..7
    const int n16  = lane & 15;
    const int kh   = lane >> 4;                // K-half selector per ISA WMMA layouts
    const int rowBlock = blockIdx.x;           // 0..127
    const int bh       = blockIdx.y;           // 0..63

    const size_t qk_base = (size_t)bh * S_ * DK_;

    // ---- A fragments (Q tile, scaled by 1/8), built once per workgroup ----
    // 16-bit A 16x32 layout: kk(e) = 32c + 16*kh + e
    const float* qrow = Q + qk_base + (size_t)(rowBlock * 16 + n16) * DK_;
    v16bf a_hi[2], a_lo[2];
    #pragma unroll
    for (int c = 0; c < 2; ++c) {
        const float* p0 = qrow + 32 * c + 16 * kh;
        const float* p1 = qrow + 32 * c + 16 * kh + 8;
        float4 t0 = ((const float4*)p0)[0];
        float4 t1 = ((const float4*)p0)[1];
        float4 t2 = ((const float4*)p1)[0];
        float4 t3 = ((const float4*)p1)[1];
        float v[16] = { t0.x,t0.y,t0.z,t0.w, t1.x,t1.y,t1.z,t1.w,
                        t2.x,t2.y,t2.z,t2.w, t3.x,t3.y,t3.z,t3.w };
        #pragma unroll
        for (int e = 0; e < 16; ++e) {
            __bf16 h, l;
            split_bf16(v[e] * SCALE_, h, l);
            a_hi[c][e] = h;
            a_lo[c][e] = l;
        }
    }

    // ---- 16 N-tiles of 16 columns per wave ----
    // B 32x16 layout: lane holds column N = lane&15; element e -> d = 32c + 16*kh + e,
    // i.e. a CONTIGUOUS 16-element run -> direct v16bf load from the bf16 planes.
    #pragma unroll 2
    for (int t = 0; t < 16; ++t) {
        const int colbase = wave * 256 + t * 16;
        v8f acc0 = {};   // hi*hi chain
        v8f acc1 = {};   // lo*hi + hi*lo chain (independent -> WMMA ILP)
        #pragma unroll
        for (int c = 0; c < 2; ++c) {
            v16bf bhv, blv;
            if constexpr (PRECONV) {
                const size_t roff = qk_base + (size_t)(colbase + n16) * DK_ + 32 * c + 16 * kh;
                bhv = *(const v16bf*)(Khi + roff);   // 32B contiguous -> 2x global_load_b128
                blv = *(const v16bf*)(Klo + roff);
            } else {
                const float* p = K + qk_base + (size_t)(colbase + n16) * DK_ + 32 * c + 16 * kh;
                float4 u0 = ((const float4*)p)[0];
                float4 u1 = ((const float4*)p)[1];
                float4 u2 = ((const float4*)p)[2];
                float4 u3 = ((const float4*)p)[3];
                float v[16] = { u0.x,u0.y,u0.z,u0.w, u1.x,u1.y,u1.z,u1.w,
                                u2.x,u2.y,u2.z,u2.w, u3.x,u3.y,u3.z,u3.w };
                #pragma unroll
                for (int e = 0; e < 16; ++e) {
                    __bf16 h, l;
                    split_bf16(v[e], h, l);
                    bhv[e] = h;
                    blv[e] = l;
                }
            }
            // 3xBF16 fp32 emulation: hi*hi + lo*hi + hi*lo (drop lo*lo)
            acc0 = __builtin_amdgcn_wmma_f32_16x16x32_bf16(false, a_hi[c], false, bhv,
                                                           (short)0, acc0, false, false);
            acc1 = __builtin_amdgcn_wmma_f32_16x16x32_bf16(false, a_lo[c], false, bhv,
                                                           (short)0, acc1, false, false);
            acc1 = __builtin_amdgcn_wmma_f32_16x16x32_bf16(false, a_hi[c], false, blv,
                                                           (short)0, acc1, false, false);
        }
        // C 16x16 layout: VGPR e, lanes 0-15 -> (M=e, N=lane); lanes 16-31 -> (M=e+8)
        #pragma unroll
        for (int e = 0; e < 8; ++e) {
            sc[(size_t)(e + 8 * kh) * SP_ + colbase + n16] = acc0[e] + acc1[e];
        }
    }

    __syncthreads();

    // ---- Row softmax: each wave owns 2 of the 16 rows ----
    #pragma unroll 1
    for (int rr = 0; rr < 2; ++rr) {
        const int r = wave * 2 + rr;
        float4* srow = (float4*)(sc + (size_t)r * SP_);

        float m = -3.402823466e38f;
        #pragma unroll 1
        for (int i = 0; i < 16; ++i) {
            float4 v = srow[lane + 32 * i];
            m = fmaxf(m, fmaxf(fmaxf(v.x, v.y), fmaxf(v.z, v.w)));
        }
        #pragma unroll
        for (int off = 16; off > 0; off >>= 1)
            m = fmaxf(m, __shfl_xor(m, off, 32));

        float s = 0.f;
        #pragma unroll 1
        for (int i = 0; i < 16; ++i) {
            float4 v = srow[lane + 32 * i];
            v.x = __expf(v.x - m); v.y = __expf(v.y - m);
            v.z = __expf(v.z - m); v.w = __expf(v.w - m);
            srow[lane + 32 * i] = v;
            s += v.x + v.y + v.z + v.w;
        }
        #pragma unroll
        for (int off = 16; off > 0; off >>= 1)
            s += __shfl_xor(s, off, 32);
        const float inv = 1.0f / s;

        float4* orow = (float4*)(Out + ((size_t)bh * S_ + rowBlock * 16 + r) * S_);
        #pragma unroll 1
        for (int i = 0; i < 16; ++i) {
            float4 v = srow[lane + 32 * i];
            v.x *= inv; v.y *= inv; v.z *= inv; v.w *= inv;
            orow[lane + 32 * i] = v;               // coalesced global_store_b128
        }
    }
}

extern "C" void kernel_launch(void* const* d_in, const int* in_sizes, int n_in,
                              void* d_out, int out_size, void* d_ws, size_t ws_size,
                              hipStream_t stream) {
    const float* Q = (const float*)d_in[0];
    const float* K = (const float*)d_in[1];
    // d_in[2] (value) is unused: reference returns softmax(Q K^T / sqrt(dk)) only.
    float* Out = (float*)d_out;

    const size_t nK   = (size_t)B_ * H_ * S_ * DK_;        // 8,388,608 elements
    const size_t need = 2 * nK * sizeof(unsigned short);   // Khi + Klo = 33.5 MB

    dim3 grid(S_ / 16, B_ * H_);   // 128 row-blocks x 64 (b,h)
    dim3 block(256);               // 8 wave32 waves
    size_t smem = (size_t)16 * SP_ * sizeof(float);        // 131,328 B LDS

    if (ws_size >= need) {
        unsigned short* Khi = (unsigned short*)d_ws;
        unsigned short* Klo = Khi + nK;
        convert_k_kernel<<<dim3((unsigned)(nK / 4 / 256)), dim3(256), 0, stream>>>(K, Khi, Klo);
        attn_softmax_kernel<true><<<grid, block, smem, stream>>>(Q, K, Khi, Klo, Out);
    } else {
        attn_softmax_kernel<false><<<grid, block, smem, stream>>>(Q, K, nullptr, nullptr, Out);
    }
}
